// GRUMessage_39092792328632
// MI455X (gfx1250) — matrix operand order repaired and checked by reading, not compiled, not run on hardware
//
#include <hip/hip_runtime.h>
#include <hip/hip_bf16.h>

// ---------------------------------------------------------------------------
// GRU message-passing, MI455X (gfx1250, wave32, WMMA bf16).
// h kept in f32 in d_out; big GEMMs on v_wmma_f32_16x16x32_bf16.
// LDS staging via GLOBAL_LOAD_ASYNC_TO_LDS_B128 (ASYNCcnt) when available.
// ---------------------------------------------------------------------------

typedef __attribute__((ext_vector_type(16))) __bf16 v16bf;
typedef __attribute__((ext_vector_type(8)))  float  v8f;
typedef __attribute__((ext_vector_type(4)))  int    v4i;

constexpr int BIMG = 16;
constexpr int RROW = 1024;
constexpr int FD   = 1024;          // feature dim == input dim
constexpr int NROW = BIMG * RROW;   // 16384
constexpr int LDA  = 40;            // padded LDS row stride (halfwords)

#if defined(__HIP_DEVICE_COMPILE__) && defined(__gfx1250__) && \
    __has_builtin(__builtin_amdgcn_global_load_async_to_lds_b128)
#define USE_ASYNC_LDS 1
#else
#define USE_ASYNC_LDS 0
#endif

#if USE_ASYNC_LDS
typedef __attribute__((address_space(1))) v4i* gptr_v4i;
typedef __attribute__((address_space(3))) v4i* lptr_v4i;
#endif

// ---- bf16 helpers (storage type = unsigned short everywhere) --------------
static __device__ __forceinline__ unsigned short f2bf(float x) {
  union { float f; unsigned u; } v; v.f = x;
  unsigned u = v.u + 0x7FFFu + ((v.u >> 16) & 1u);   // round-to-nearest-even
  return (unsigned short)(u >> 16);
}

// 16-byte global -> LDS copy: async DMA path (ASYNCcnt) or VGPR fallback.
static __device__ __forceinline__ void cp16_g2l(const unsigned short* g,
                                                unsigned short* l) {
#if USE_ASYNC_LDS
  __builtin_amdgcn_global_load_async_to_lds_b128(
      (gptr_v4i)(unsigned short*)g, (lptr_v4i)l, 0, 0);
#else
  *(uint4*)l = *(const uint4*)g;
#endif
}

static __device__ __forceinline__ void cp_wait() {
#if USE_ASYNC_LDS
#if __has_builtin(__builtin_amdgcn_s_wait_asynccnt)
  __builtin_amdgcn_s_wait_asynccnt(0);
#else
  asm volatile("s_wait_asynccnt 0x0" ::: "memory");
#endif
#endif
}

union FragBF { v16bf v; uint4 q[2]; };

// A fragment 16x32 bf16 (ISA 7.12.2): lane l: m=l&15, K-half=(l>>4)*8,
// elements t0..7 -> K = kh+t, t8..15 -> K = 16+kh+(t-8). Two 16B LDS reads.
static __device__ __forceinline__ v16bf frag_a(const unsigned short* base,
                                               int row0, int stride) {
  const int l  = threadIdx.x & 31;
  const int m  = l & 15;
  const int kh = (l >> 4) << 3;
  const unsigned short* p = base + (row0 + m) * stride + kh;
  FragBF f;
  f.q[0] = *(const uint4*)(p);
  f.q[1] = *(const uint4*)(p + 16);
  return f.v;
}

// B fragment 32x16 bf16: lane l: n=l&15, K-half=(l>>4)*8, t0..7 -> K = kh+t, t8..15 -> K = 16+kh+(t-8).
static __device__ __forceinline__ v16bf frag_b(const unsigned short* base,
                                               int col0, int stride) {
  const int l  = threadIdx.x & 31;
  const int n  = l & 15;
  const int ks = (l >> 4) << 3;
  const unsigned short* p = base + (col0 + n) * stride + ks;
  FragBF f;
  f.q[0] = *(const uint4*)(p);
  f.q[1] = *(const uint4*)(p + 16);
  return f.v;
}

static __device__ __forceinline__ v8f wmma_bf16(v16bf a, v16bf b, v8f c) {
  return __builtin_amdgcn_wmma_f32_16x16x32_bf16(false, a, false, b,
                                                 (short)0, c, false, false);
}

static __device__ __forceinline__ float sigmoidf_(float x) {
  return 1.0f / (1.0f + __expf(-x));
}

// ---------------------------------------------------------------------------
// Elementwise / prep kernels
// ---------------------------------------------------------------------------
__global__ __launch_bounds__(256) void k_cvt_bf16(const float* __restrict__ in,
                                                  unsigned short* __restrict__ out,
                                                  int n) {
  int id = blockIdx.x * 256 + threadIdx.x;
  if (id < n) out[id] = f2bf(in[id]);
}

// box_feat[n,f] = boxes[n,:4] . fc_box_w[f,:4] + fc_box_b[f]   (K=4 GEMM)
__global__ __launch_bounds__(256) void k_boxfeat(const float* __restrict__ boxes,
                                                 const float* __restrict__ w,
                                                 const float* __restrict__ b,
                                                 float* __restrict__ out) {
  int id = blockIdx.x * 256 + threadIdx.x;       // NROW*FD grid, exact
  int n  = id >> 10;
  int f  = id & (FD - 1);
  const float* br = boxes + n * 4;
  const float* wr = w + f * 4;
  float acc = b[f];
  acc = fmaf(br[0], wr[0], acc);
  acc = fmaf(br[1], wr[1], acc);
  acc = fmaf(br[2], wr[2], acc);
  acc = fmaf(br[3], wr[3], acc);
  out[id] = acc;
}

// y = bf16(relu(h*box_feat)); hbf = bf16(h)
__global__ __launch_bounds__(256) void k_prep(const float* __restrict__ h,
                                              const float* __restrict__ box_feat,
                                              unsigned short* __restrict__ ybf,
                                              unsigned short* __restrict__ hbf) {
  int id = blockIdx.x * 256 + threadIdx.x;
  float hv = h[id];
  float y  = fmaxf(hv * box_feat[id], 0.0f);
  ybf[id] = f2bf(y);
  hbf[id] = f2bf(hv);
}

// colsum[b,i] = sum_r x[b*R + r, i]
__global__ __launch_bounds__(256) void k_colsum(const float* __restrict__ x,
                                                float* __restrict__ colsum) {
  int id = blockIdx.x * 256 + threadIdx.x;       // BIMG*FD
  int b  = id >> 10;
  int i  = id & (FD - 1);
  const float* p = x + (size_t)b * RROW * FD + i;
  float s = 0.0f;
  for (int r = 0; r < RROW; ++r) s += p[(size_t)r * FD];
  colsum[id] = s;
}

// inp = bf16((colsum[b] - x) / (R-1))
__global__ __launch_bounds__(256) void k_inp(const float* __restrict__ x,
                                             const float* __restrict__ colsum,
                                             unsigned short* __restrict__ inp) {
  int id  = blockIdx.x * 256 + threadIdx.x;      // NROW*FD
  int row = id >> 10;
  int b   = row >> 10;
  int i   = id & (FD - 1);
  float v = (colsum[b * FD + i] - x[id]) * (1.0f / 1023.0f);
  inp[id] = f2bf(v);
}

// ---------------------------------------------------------------------------
// GEMM: x[N,FD] = ybf[N,FD] @ wfc[FD,FD]^T + bias
// block = 128x128 tile, 8 waves (4x2) each 32x64 via 8 WMMA accumulators.
// ---------------------------------------------------------------------------
__global__ __launch_bounds__(256) void k_gemm_x(const unsigned short* __restrict__ ybf,
                                                const unsigned short* __restrict__ wfc,
                                                const float* __restrict__ bias,
                                                float* __restrict__ xout) {
  __shared__ alignas(16) unsigned short sA[128 * LDA];
  __shared__ alignas(16) unsigned short sB[128 * LDA];
  const int t    = threadIdx.x;
  const int lane = t & 31;
  const int wave = t >> 5;
  const int wy   = wave >> 1;          // 0..3 -> 32-row sub-tile
  const int wx   = wave & 1;           // 0..1 -> 64-col sub-tile
  const int row0 = blockIdx.x * 128;
  const int col0 = blockIdx.y * 128;

  v8f acc[2][4];
#pragma unroll
  for (int i = 0; i < 2; ++i)
#pragma unroll
    for (int j = 0; j < 4; ++j) acc[i][j] = {};

  for (int k0 = 0; k0 < FD; k0 += 32) {
    __syncthreads();                              // WAR on LDS
#pragma unroll
    for (int u0 = 0; u0 < 2; ++u0) {              // A: 128x32 -> 512 uint4
      int u = t + u0 * 256;
      int r = u >> 2, ch = u & 3;
      cp16_g2l(&ybf[(size_t)(row0 + r) * FD + k0 + ch * 8],
               &sA[r * LDA + ch * 8]);
    }
#pragma unroll
    for (int u0 = 0; u0 < 2; ++u0) {              // B: 128x32 -> 512 uint4
      int u = t + u0 * 256;
      int r = u >> 2, ch = u & 3;
      cp16_g2l(&wfc[(size_t)(col0 + r) * FD + k0 + ch * 8],
               &sB[r * LDA + ch * 8]);
    }
    if (k0 + 32 < FD) {                           // prefetch next A slab
      __builtin_prefetch(&ybf[(size_t)(row0 + (t >> 2)) * FD + k0 + 32 + (t & 3) * 8], 0, 1);
    }
    cp_wait();                                    // this wave's copies in LDS
    __syncthreads();                              // all waves' copies visible

    v16bf a0 = frag_a(sA, wy * 32,      LDA);
    v16bf a1 = frag_a(sA, wy * 32 + 16, LDA);
#pragma unroll
    for (int j = 0; j < 4; ++j) {
      v16bf bj = frag_b(sB, wx * 64 + j * 16, LDA);
      acc[0][j] = wmma_bf16(a0, bj, acc[0][j]);
      acc[1][j] = wmma_bf16(a1, bj, acc[1][j]);
    }
  }

  // C/D layout: element e -> row = 8*(lane>>4)+e, col = lane&15
  const int rl = (lane >> 4) << 3;
  const int cl = lane & 15;
#pragma unroll
  for (int i = 0; i < 2; ++i) {
    int rbase = row0 + wy * 32 + i * 16 + rl;
#pragma unroll
    for (int j = 0; j < 4; ++j) {
      int c    = col0 + wx * 64 + j * 16 + cl;
      float bv = bias[c];
#pragma unroll
      for (int e = 0; e < 8; ++e)
        xout[(size_t)(rbase + e) * FD + c] = acc[i][j][e] + bv;
    }
  }
}

// ---------------------------------------------------------------------------
// Fused GRU: per 64x32 output tile compute 6 gate GEMMs
//   ai[g] = inp @ w_ih[g]^T , ah[g] = hbf @ w_hh[g]^T   (g = r,z,n)
// then gates + in-place f32 h update (gh reads the bf16 snapshot, so safe).
// ---------------------------------------------------------------------------
__global__ __launch_bounds__(256) void k_gru(const unsigned short* __restrict__ inp,
                                             const unsigned short* __restrict__ hbf,
                                             const unsigned short* __restrict__ wih,
                                             const unsigned short* __restrict__ whh,
                                             const float* __restrict__ b_ih,
                                             const float* __restrict__ b_hh,
                                             float* __restrict__ h) {
  __shared__ alignas(16) unsigned short sI[64 * LDA];
  __shared__ alignas(16) unsigned short sH[64 * LDA];
  __shared__ alignas(16) unsigned short sW[6 * 32 * LDA];
  const int t    = threadIdx.x;
  const int lane = t & 31;
  const int wave = t >> 5;
  const int wy   = wave >> 1;          // 0..3 -> 16-row sub-tile
  const int wx   = wave & 1;           // 0..1 -> 16-col sub-tile
  const int row0 = blockIdx.x * 64;
  const int col0 = blockIdx.y * 32;

  v8f ai0 = {}, ai1 = {}, ai2 = {}, ah0 = {}, ah1 = {}, ah2 = {};

  for (int k0 = 0; k0 < FD; k0 += 32) {
    __syncthreads();
    {                                             // inp & h tiles: 64x32 each
      int r = t >> 2, ch = t & 3;
      cp16_g2l(&inp[(size_t)(row0 + r) * FD + k0 + ch * 8],
               &sI[r * LDA + ch * 8]);
      cp16_g2l(&hbf[(size_t)(row0 + r) * FD + k0 + ch * 8],
               &sH[r * LDA + ch * 8]);
    }
#pragma unroll
    for (int u0 = 0; u0 < 3; ++u0) {              // 6 weight tiles 32x32
      int u    = t + u0 * 256;
      int g3   = u >> 7;                          // 0..5
      int r    = (u >> 2) & 31;
      int ch   = u & 3;
      int gate = (g3 < 3) ? g3 : g3 - 3;
      const unsigned short* W = (g3 < 3) ? wih : whh;
      cp16_g2l(&W[(size_t)(gate * FD + col0 + r) * FD + k0 + ch * 8],
               &sW[g3 * 32 * LDA + r * LDA + ch * 8]);
    }
    if (k0 + 32 < FD) {
      __builtin_prefetch(&inp[(size_t)(row0 + (t >> 2)) * FD + k0 + 32 + (t & 3) * 8], 0, 1);
      __builtin_prefetch(&hbf[(size_t)(row0 + (t >> 2)) * FD + k0 + 32 + (t & 3) * 8], 0, 1);
    }
    cp_wait();
    __syncthreads();

    v16bf fai = frag_a(sI, wy * 16, LDA);
    v16bf fah = frag_a(sH, wy * 16, LDA);
    ai0 = wmma_bf16(fai, frag_b(&sW[0 * 32 * LDA], wx * 16, LDA), ai0);
    ah0 = wmma_bf16(fah, frag_b(&sW[3 * 32 * LDA], wx * 16, LDA), ah0);
    ai1 = wmma_bf16(fai, frag_b(&sW[1 * 32 * LDA], wx * 16, LDA), ai1);
    ah1 = wmma_bf16(fah, frag_b(&sW[4 * 32 * LDA], wx * 16, LDA), ah1);
    ai2 = wmma_bf16(fai, frag_b(&sW[2 * 32 * LDA], wx * 16, LDA), ai2);
    ah2 = wmma_bf16(fah, frag_b(&sW[5 * 32 * LDA], wx * 16, LDA), ah2);
  }

  const int rl = (lane >> 4) << 3;
  const int cl = lane & 15;
  const int c  = col0 + wx * 16 + cl;
  const float bir = b_ih[c],          bhr = b_hh[c];
  const float biz = b_ih[FD + c],     bhz = b_hh[FD + c];
  const float bin = b_ih[2 * FD + c], bhn = b_hh[2 * FD + c];
  const int rbase = row0 + wy * 16 + rl;
#pragma unroll
  for (int j = 0; j < 8; ++j) {
    float r = sigmoidf_((ai0[j] + bir) + (ah0[j] + bhr));
    float z = sigmoidf_((ai1[j] + biz) + (ah1[j] + bhz));
    float n = tanhf((ai2[j] + bin) + r * (ah2[j] + bhn));
    size_t idx = (size_t)(rbase + j) * FD + c;
    float hold = h[idx];
    h[idx] = (1.0f - z) * n + z * hold;
  }
}

// ---------------------------------------------------------------------------
// Host orchestration
// ---------------------------------------------------------------------------
extern "C" void kernel_launch(void* const* d_in, const int* in_sizes, int n_in,
                              void* d_out, int out_size, void* d_ws, size_t ws_size,
                              hipStream_t stream) {
  const float* features   = (const float*)d_in[0];
  const float* boxes      = (const float*)d_in[1];
  const float* fc_box_w   = (const float*)d_in[2];
  const float* fc_box_b   = (const float*)d_in[3];
  const float* fc_input_w = (const float*)d_in[4];
  const float* fc_input_b = (const float*)d_in[5];
  const float* w_ih       = (const float*)d_in[6];
  const float* w_hh       = (const float*)d_in[7];
  const float* b_ih       = (const float*)d_in[8];
  const float* b_hh       = (const float*)d_in[9];
  float* h = (float*)d_out;                         // [NROW, FD] f32

  // Workspace carving (256B aligned segments)
  unsigned char* w = (unsigned char*)d_ws;
  auto carve = [&](size_t bytes) {
    void* p = (void*)w;
    w += (bytes + 255) & ~(size_t)255;
    return p;
  };
  float*          box_feat = (float*)carve((size_t)NROW * FD * 4);
  float*          xbuf     = (float*)carve((size_t)NROW * FD * 4);
  float*          colsum   = (float*)carve((size_t)BIMG * FD * 4);
  unsigned short* ybf      = (unsigned short*)carve((size_t)NROW * FD * 2);
  unsigned short* hbf      = (unsigned short*)carve((size_t)NROW * FD * 2);
  unsigned short* inpbf    = (unsigned short*)carve((size_t)NROW * FD * 2);
  unsigned short* wfc_bf   = (unsigned short*)carve((size_t)FD * FD * 2);
  unsigned short* wih_bf   = (unsigned short*)carve((size_t)3 * FD * FD * 2);
  unsigned short* whh_bf   = (unsigned short*)carve((size_t)3 * FD * FD * 2);
  (void)ws_size; (void)n_in; (void)in_sizes; (void)out_size;

  const int NF = NROW * FD;            // 16,777,216

  // h <- features
  (void)hipMemcpyAsync(h, features, (size_t)NF * sizeof(float),
                       hipMemcpyDeviceToDevice, stream);

  // Weights -> bf16 (once)
  k_cvt_bf16<<<(FD * FD) / 256, 256, 0, stream>>>(fc_input_w, wfc_bf, FD * FD);
  k_cvt_bf16<<<(3 * FD * FD) / 256, 256, 0, stream>>>(w_ih, wih_bf, 3 * FD * FD);
  k_cvt_bf16<<<(3 * FD * FD) / 256, 256, 0, stream>>>(w_hh, whh_bf, 3 * FD * FD);

  // box_feat (hoisted, K=4)
  k_boxfeat<<<NF / 256, 256, 0, stream>>>(boxes, fc_box_w, fc_box_b, box_feat);

  dim3 gx(NROW / 128, FD / 128);       // 128 x 8 blocks
  dim3 gg(NROW / 64, FD / 32);         // 256 x 32 blocks

  for (int it = 0; it < 2; ++it) {
    k_prep<<<NF / 256, 256, 0, stream>>>(h, box_feat, ybf, hbf);
    k_gemm_x<<<gx, 256, 0, stream>>>(ybf, wfc_bf, fc_input_b, xbuf);
    k_colsum<<<(BIMG * FD) / 256, 256, 0, stream>>>(xbuf, colsum);
    k_inp<<<NF / 256, 256, 0, stream>>>(xbuf, colsum, inpbf);
    k_gru<<<gg, 256, 0, stream>>>(inpbf, hbf, wih_bf, whh_bf, b_ih, b_hh, h);
  }
}
